// MultiHeadBatched_74071005987473
// MI455X (gfx1250) — matrix operand order repaired and checked by reading, not compiled, not run on hardware
//
#include <hip/hip_runtime.h>
#include <hip/hip_bf16.h>
#include <math.h>
#include <stdint.h>

typedef __attribute__((ext_vector_type(16))) _Float16 v16h;
typedef __attribute__((ext_vector_type(8)))  float    v8f;
typedef __attribute__((ext_vector_type(4)))  unsigned int u32x4;
typedef __attribute__((ext_vector_type(8)))  int          i32x8;
typedef __attribute__((ext_vector_type(4)))  int          i32x4;

#define EMB      1024
#define HEADS    16
#define HEAD_DIM 64
#define BATCH    2
#define SEQ      2048

// --- A-fragment K index for 16-bit 16x32 A matrix (contiguous k order, same as B operands) ---
// VGPR pair p (0..7): k = 2p + (lane>=16 ? 16 : 0) (+0/+1)
__device__ __forceinline__ int a_pair_k(int p, int hi) {
    return p * 2 + (hi ? 16 : 0);
}

// ---------------------------------------------------------------------------
// Tensor Data Mover: 2D f16 tile (rows x cols) from global -> LDS.
// Descriptor packing per CDNA5 ISA.
// Issue once per workgroup (wave 0); completion via s_wait_tensorcnt.
// 6-arg builtin form (clang-23 / therock-10.0 toolchain).
// ---------------------------------------------------------------------------
__device__ __forceinline__ void tdm_load_tile_f16(unsigned lds_byte_addr,
                                                  const _Float16* gptr,
                                                  int rows, int cols,
                                                  int row_stride_elems)
{
    const unsigned long long ga = (unsigned long long)(uintptr_t)gptr;
    u32x4 g0;
    g0[0] = 1u;                                        // count=1 (valid), no gather
    g0[1] = lds_byte_addr;                             // D#.lds_addr
    g0[2] = (unsigned)(ga & 0xffffffffu);              // global_addr[31:0]
    g0[3] = (unsigned)((ga >> 32) & 0x1ffffffu)        // global_addr[56:32]
          | (2u << 30);                                // type=2 ("image")
    i32x8 g1;
    g1[0] = (1 << 16);                                 // data_size=1 (2B), wg_mask=0
    g1[1] = (cols & 0xffff) << 16;                     // tensor_dim0[15:0]
    g1[2] = ((cols >> 16) & 0xffff)                    // tensor_dim0[31:16]
          | ((rows & 0xffff) << 16);                   // tensor_dim1[15:0]
    g1[3] = ((rows >> 16) & 0xffff)                    // tensor_dim1[31:16]
          | ((cols & 0xffff) << 16);                   // tile_dim0 = cols
    g1[4] = (rows & 0xffff);                           // tile_dim1 = rows, tile_dim2=0
    g1[5] = row_stride_elems;                          // tensor_dim0_stride[31:0]
    g1[6] = 0;                                         // stride[47:32]=0, dim1_stride=0
    g1[7] = 0;
    const i32x4 z4 = {};
    const i32x8 z8 = {};
    __builtin_amdgcn_tensor_load_to_lds(g0, g1, z4, z4, z8, 0);
}

// ---------------------------------------------------------------------------
// C[m,n] = sum_k A[m,k] * W[n,k] + bias[n]     (A @ W^T + b)
// block = 128 threads (4 waves); wave w owns rows [blockIdx.x*64 + 16w, +16),
// cols [blockIdx.y*64, +64) as 4 C fragments (A fragment reused 4x per K step).
// ---------------------------------------------------------------------------
template <bool A_F32, bool OUT_F32>
__global__ __launch_bounds__(128)
void gemm_bias_kernel(const void* __restrict__ Aptr,
                      const float* __restrict__ W,
                      const float* __restrict__ bias,
                      void* __restrict__ Outptr,
                      int M, int N, int K)
{
    const int lane = threadIdx.x & 31;
    const int wave = threadIdx.x >> 5;
    const int hi   = lane >> 4;
    const int ln   = lane & 15;

    const int m0 = blockIdx.x * 64 + wave * 16;
    const int n0 = blockIdx.y * 64;

    const float*    Af = (const float*)Aptr;
    const _Float16* Ah = (const _Float16*)Aptr;

    const int mrow = m0 + ln;

    v8f c[4] = {};

    for (int kb = 0; kb < K; kb += 32) {
        // Prefetch next A chunk for this row
        if (kb + 32 < K) {
            const size_t next = (size_t)mrow * K + kb + 32;
            __builtin_prefetch((const char*)Aptr + next * (A_F32 ? 4 : 2), 0, 3);
        }
        // A fragment: 16x32 f16
        v16h a;
#pragma unroll
        for (int p = 0; p < 8; ++p) {
            const int k = kb + a_pair_k(p, hi);
            if constexpr (A_F32) {
                a[2 * p]     = (_Float16)Af[(size_t)mrow * K + k];
                a[2 * p + 1] = (_Float16)Af[(size_t)mrow * K + k + 1];
            } else {
                a[2 * p]     = Ah[(size_t)mrow * K + k];
                a[2 * p + 1] = Ah[(size_t)mrow * K + k + 1];
            }
        }
        // 4 B fragments (32x16 each): B[k][n] = W[n][k]; contiguous k-run of 16/lane
#pragma unroll
        for (int nt = 0; nt < 4; ++nt) {
            v16h b;
            const float* wrow = &W[(size_t)(n0 + nt * 16 + ln) * K + kb + hi * 16];
#pragma unroll
            for (int e = 0; e < 16; ++e) b[e] = (_Float16)wrow[e];
            c[nt] = __builtin_amdgcn_wmma_f32_16x16x32_f16(false, a, false, b,
                                                           (short)0, c[nt],
                                                           false, false);
        }
    }

#pragma unroll
    for (int nt = 0; nt < 4; ++nt) {
        const int ncol = n0 + nt * 16 + ln;
        const float bn = bias[ncol];
        if constexpr (OUT_F32) {
            float* Of = (float*)Outptr;
#pragma unroll
            for (int r = 0; r < 8; ++r)
                Of[(size_t)(m0 + r + hi * 8) * N + ncol] = c[nt][r] + bn;
        } else {
            _Float16* Oh = (_Float16*)Outptr;
#pragma unroll
            for (int r = 0; r < 8; ++r)
                Oh[(size_t)(m0 + r + hi * 8) * N + ncol] = (_Float16)(c[nt][r] + bn);
        }
    }
}

// ---------------------------------------------------------------------------
// Flash attention over one (b,h), 64 query rows per block (16 per wave).
// K tile staged via TDM (tensor_load_to_lds); V tile staged transposed so the
// P@V B-fragment is a single contiguous 32-byte LDS read.
// ---------------------------------------------------------------------------
__global__ __launch_bounds__(128)
void flash_attn_kernel(const _Float16* __restrict__ Qh,
                       const _Float16* __restrict__ Kh,
                       const _Float16* __restrict__ Vh,
                       const int* __restrict__ mask,
                       _Float16* __restrict__ Ctx)
{
    __shared__ __align__(32) _Float16 lds_k[32 * HEAD_DIM];    // (key,d) 4 KB (TDM)
    __shared__ __align__(32) _Float16 lds_vt[HEAD_DIM * 32];   // (d,key) 4 KB
    __shared__ __align__(32) _Float16 lds_p[4][16 * 32];       // per-wave P 4 KB

    const int lane = threadIdx.x & 31;
    const int wave = threadIdx.x >> 5;
    const int hi   = lane >> 4;
    const int ln   = lane & 15;
    const int tid  = threadIdx.x;

    const int qtiles = SEQ / 64;
    const int qt = blockIdx.x % qtiles;
    const int bh = blockIdx.x / qtiles;
    const int b  = bh / HEADS;
    const int h  = bh % HEADS;

    const size_t base = ((size_t)b * SEQ) * EMB + (size_t)h * HEAD_DIM;
    const int m0 = qt * 64 + wave * 16;
    const float scale = 0.125f;   // HEAD_DIM^-0.5

    const unsigned ldsK_addr = (unsigned)(uintptr_t)(void*)&lds_k[0];

    // Load Q A-fragments once: 16x64 = two 16x32 fragments
    v16h aq[2];
    const _Float16* qrow = &Qh[base + (size_t)(m0 + ln) * EMB];
#pragma unroll
    for (int kc = 0; kc < 2; ++kc) {
#pragma unroll
        for (int p = 0; p < 8; ++p) {
            const int k = kc * 32 + a_pair_k(p, hi);
            aq[kc][2 * p]     = qrow[k];
            aq[kc][2 * p + 1] = qrow[k + 1];
        }
    }

    v8f o[4] = {};
    float rmax[8], rsum[8];
#pragma unroll
    for (int r = 0; r < 8; ++r) { rmax[r] = -INFINITY; rsum[r] = 0.0f; }

    for (int kb = 0; kb < SEQ; kb += 32) {
        __syncthreads();   // previous iteration's LDS consumers done
        // --- K tile via Tensor Data Mover (wave 0 issues; EXEC-independent) ---
        if (wave == 0) {
            tdm_load_tile_f16(ldsK_addr, &Kh[base + (size_t)kb * EMB],
                              32, HEAD_DIM, EMB);
        }
        // --- V tile: cooperative load, store transposed (d, key) ---
        {
            const int key = tid >> 2;            // 0..31
            const int dv  = (tid & 3) * 16;      // 0,16,32,48
            const v16h vv = *(const v16h*)&Vh[base + (size_t)(kb + key) * EMB + dv];
#pragma unroll
            for (int e = 0; e < 16; ++e)
                lds_vt[(dv + e) * 32 + key] = vv[e];
        }
        if (wave == 0) __builtin_amdgcn_s_wait_tensorcnt(0);
        __syncthreads();

        // Scores: S = Q @ K^T, two 16-key n-subtiles, K-dim 64 = 2 x 32
        v8f s[2];
#pragma unroll
        for (int nt = 0; nt < 2; ++nt) {
            v8f acc = {};
#pragma unroll
            for (int kc = 0; kc < 2; ++kc) {
                const v16h bk = *(const v16h*)
                    &lds_k[(nt * 16 + ln) * HEAD_DIM + kc * 32 + hi * 16];
                acc = __builtin_amdgcn_wmma_f32_16x16x32_f16(
                    false, aq[kc], false, bk, (short)0, acc, false, false);
            }
            s[nt] = acc;
        }

        // Scale + mask
#pragma unroll
        for (int nt = 0; nt < 2; ++nt) {
            const bool ok = mask[b * SEQ + kb + nt * 16 + ln] != 0;
#pragma unroll
            for (int r = 0; r < 8; ++r) {
                const float sv = s[nt][r] * scale;
                s[nt][r] = ok ? sv : -INFINITY;
            }
        }

        // Online softmax (row = 16 contiguous lanes of one half-wave)
        float nmax[8], corr[8];
#pragma unroll
        for (int r = 0; r < 8; ++r) {
            float m = fmaxf(s[0][r], s[1][r]);
#pragma unroll
            for (int off = 1; off < 16; off <<= 1)
                m = fmaxf(m, __shfl_xor(m, off, 32));
            const float nm = fmaxf(rmax[r], m);
            corr[r] = (nm == -INFINITY) ? 1.0f : __expf(rmax[r] - nm);
            nmax[r] = nm;
        }
#pragma unroll
        for (int nt = 0; nt < 2; ++nt)
#pragma unroll
            for (int r = 0; r < 8; ++r) {
                const float sv = s[nt][r];
                s[nt][r] = (sv == -INFINITY) ? 0.0f : __expf(sv - nmax[r]);
            }
#pragma unroll
        for (int r = 0; r < 8; ++r) {
            float sum = s[0][r] + s[1][r];
#pragma unroll
            for (int off = 1; off < 16; off <<= 1)
                sum += __shfl_xor(sum, off, 32);
            rsum[r] = rsum[r] * corr[r] + sum;
            rmax[r] = nmax[r];
        }
#pragma unroll
        for (int j = 0; j < 4; ++j)
#pragma unroll
            for (int r = 0; r < 8; ++r) o[j][r] *= corr[r];

        // Transpose P (C-layout) -> A-layout through per-wave LDS scratch
        _Float16* pb = lds_p[wave];
#pragma unroll
        for (int nt = 0; nt < 2; ++nt)
#pragma unroll
            for (int r = 0; r < 8; ++r)
                pb[(r + hi * 8) * 32 + nt * 16 + ln] = (_Float16)s[nt][r];
        asm volatile("s_wait_dscnt 0" ::: "memory");   // wave-local DS RAW

        v16h ap;
#pragma unroll
        for (int p = 0; p < 8; ++p) {
            const int k = a_pair_k(p, hi);
            ap[2 * p]     = pb[ln * 32 + k];
            ap[2 * p + 1] = pb[ln * 32 + k + 1];
        }

        // O += P @ V  : bv[e] = V[key=hi*16+e][d=j*16+ln] = lds_vt[d*32 + key]
#pragma unroll
        for (int j = 0; j < 4; ++j) {
            const v16h bv = *(const v16h*)&lds_vt[(j * 16 + ln) * 32 + hi * 16];
            o[j] = __builtin_amdgcn_wmma_f32_16x16x32_f16(
                false, ap, false, bv, (short)0, o[j], false, false);
        }
    }

    // Normalize and store context (f16, [B][S][H][D])
#pragma unroll
    for (int r = 0; r < 8; ++r) {
        const float inv = (rsum[r] > 0.0f) ? (1.0f / rsum[r]) : 0.0f;
        const int row = m0 + r + hi * 8;
#pragma unroll
        for (int j = 0; j < 4; ++j)
            Ctx[base + (size_t)row * EMB + j * 16 + ln] = (_Float16)(o[j][r] * inv);
    }
}

// ---------------------------------------------------------------------------
extern "C" void kernel_launch(void* const* d_in, const int* in_sizes, int n_in,
                              void* d_out, int out_size, void* d_ws, size_t ws_size,
                              hipStream_t stream)
{
    const float* q    = (const float*)d_in[0];
    const float* k    = (const float*)d_in[1];
    const float* v    = (const float*)d_in[2];
    const int*   mask = (const int*)d_in[3];
    const float* Wq   = (const float*)d_in[4];
    const float* bq   = (const float*)d_in[5];
    const float* Wk   = (const float*)d_in[6];
    const float* bk   = (const float*)d_in[7];
    const float* Wv   = (const float*)d_in[8];
    const float* bv   = (const float*)d_in[9];
    const float* Wo   = (const float*)d_in[10];
    const float* bo   = (const float*)d_in[11];

    const int M = BATCH * SEQ;                 // 4096 rows
    _Float16* Qh  = (_Float16*)d_ws;
    _Float16* Kh  = Qh + (size_t)M * EMB;
    _Float16* Vh  = Kh + (size_t)M * EMB;
    _Float16* Ctx = Vh + (size_t)M * EMB;      // 32 MB total f16 scratch

    dim3 gblk(128);
    dim3 ggrid(M / 64, EMB / 64);

    // Q/K/V projections (fp32 in -> f16 out)
    gemm_bias_kernel<true, false><<<ggrid, gblk, 0, stream>>>(q, Wq, bq, Qh, M, EMB, EMB);
    gemm_bias_kernel<true, false><<<ggrid, gblk, 0, stream>>>(k, Wk, bk, Kh, M, EMB, EMB);
    gemm_bias_kernel<true, false><<<ggrid, gblk, 0, stream>>>(v, Wv, bv, Vh, M, EMB, EMB);

    // Flash attention: one block per (b, h, 64-row q tile)
    dim3 fgrid(BATCH * HEADS * (SEQ / 64));
    flash_attn_kernel<<<fgrid, gblk, 0, stream>>>(Qh, Kh, Vh, mask, Ctx);

    // Output projection (f16 in -> fp32 out)
    gemm_bias_kernel<false, true><<<ggrid, gblk, 0, stream>>>(Ctx, Wo, bo, d_out, M, EMB, EMB);
}
